// ConvLSTMCell_48112223650010
// MI455X (gfx1250) — matrix operand order repaired and checked by reading, not compiled, not run on hardware
//
#include <hip/hip_runtime.h>
#include <hip/hip_bf16.h>

// ---------------------------------------------------------------------------
// Deformable hypernet ConvLSTM cell for MI455X (gfx1250, wave32, WMMA).
// Dominant cost: per-batch GEMM M=256 N=4096 K=1152 (9.7 GFLOP) -> bf16 WMMA,
// double-buffered LDS staging, async global->LDS when the toolchain has it.
// ---------------------------------------------------------------------------

typedef __attribute__((ext_vector_type(16))) __bf16 v16bf;
typedef __attribute__((ext_vector_type(8)))  float  v8f;
typedef __attribute__((ext_vector_type(4)))  int    v4i;

#define Bm   4
#define CIN  64
#define HID  64
#define Hh   64
#define Ww   64
#define INC  128
#define OUTC 256
#define NN   9
#define Dd   256
#define KK   1152          // INC * NN
#define HW   4096          // Hh * Ww
#define HP   66            // padded height/width
#define NKC  (KK / 32)     // 36 k-chunks

#if defined(__has_builtin)
#if __has_builtin(__builtin_amdgcn_global_load_async_to_lds_b128) && \
    __has_builtin(__builtin_amdgcn_s_wait_asynccnt)
#define USE_ASYNC_LDS 1
typedef __attribute__((address_space(1))) v4i* gptr_v4i;   // global AS int4*
typedef __attribute__((address_space(3))) v4i* lptr_v4i;   // LDS AS int4*
#endif
#endif

__device__ __forceinline__ unsigned short f2bf(float f) {
    union { float f; unsigned u; } x; x.f = f;
    unsigned r = x.u + 0x7FFFu + ((x.u >> 16) & 1u);   // round-to-nearest-even
    return (unsigned short)(r >> 16);
}

__device__ __forceinline__ float sigmoidf(float x) {
    return 1.0f / (1.0f + __expf(-x));
}

// ---------------------------------------------------------------------------
// K1: wm1[b][i] = meta[b,:] . w1_w[i,:] + w1_b[i]      (4 x 32768, K=256)
// ---------------------------------------------------------------------------
__global__ __launch_bounds__(256) void hyper_w1(const float* __restrict__ meta,
                                                const float* __restrict__ w1w,
                                                const float* __restrict__ w1b,
                                                float* __restrict__ wm1) {
    int idx = blockIdx.x * 256 + threadIdx.x;          // 131072 total
    int b = idx >> 15, i = idx & 32767;
    const float4* wr = (const float4*)(w1w + (size_t)i * Dd);
    const float4* mr = (const float4*)(meta + (size_t)b * Dd);
    float acc = w1b[i];
#pragma unroll 8
    for (int d = 0; d < Dd / 4; ++d) {
        float4 a = mr[d], w = wr[d];
        acc += a.x * w.x + a.y * w.y + a.z * w.z + a.w * w.w;
    }
    wm1[idx] = acc;
}

// ---------------------------------------------------------------------------
// K2: per-sample conv weights -> bf16 WMMA A-matrix  A[b][o][c*9+n]
// ---------------------------------------------------------------------------
__global__ __launch_bounds__(256) void hyper_w2(const float* __restrict__ wm1,
                                                const float* __restrict__ w2w,
                                                const float* __restrict__ w2b,
                                                unsigned short* __restrict__ Abf) {
    int idx = blockIdx.x * 256 + threadIdx.x;          // 4*128*2304 = 1179648
    int b   = idx / (INC * OUTC * NN);
    int r   = idx % (INC * OUTC * NN);
    int c   = r / (OUTC * NN);
    int oid = r % (OUTC * NN);                          // o*9 + n
    int o = oid / NN, n = oid % NN;
    const float4* xr = (const float4*)(wm1 + ((size_t)b * INC + c) * Dd);
    const float4* wr = (const float4*)(w2w + (size_t)oid * Dd);
    float acc = w2b[oid];
#pragma unroll 8
    for (int d = 0; d < Dd / 4; ++d) {
        float4 a = xr[d], w = wr[d];
        acc += a.x * w.x + a.y * w.y + a.z * w.z + a.w * w.w;
    }
    Abf[((size_t)b * OUTC + o) * KK + (size_t)c * NN + n] = f2bf(acc);
}

// ---------------------------------------------------------------------------
// K3: gate bias  bm[b][o] = meta[b,:] . bl_w[o,:] + bl_b[o]
// ---------------------------------------------------------------------------
__global__ __launch_bounds__(256) void hyper_bias(const float* __restrict__ meta,
                                                  const float* __restrict__ blw,
                                                  const float* __restrict__ blb,
                                                  float* __restrict__ bmv) {
    int idx = blockIdx.x * 256 + threadIdx.x;          // 1024
    int b = idx >> 8, o = idx & 255;
    const float4* wr = (const float4*)(blw + (size_t)o * Dd);
    const float4* mr = (const float4*)(meta + (size_t)b * Dd);
    float acc = blb[o];
#pragma unroll 8
    for (int d = 0; d < Dd / 4; ++d) {
        float4 a = mr[d], w = wr[d];
        acc += a.x * w.x + a.y * w.y + a.z * w.z + a.w * w.w;
    }
    bmv[idx] = acc;
}

// ---------------------------------------------------------------------------
// K4: fused offset-conv + deformable bilinear sampling.
//     One block per (b,h,w) pixel, 128 threads = one per input channel.
//     Writes x_off as bf16 WMMA B-matrix:  Bbf[b][k=c*9+n][hw]
// ---------------------------------------------------------------------------
__device__ __forceinline__ float fetch_pad(const float* __restrict__ src, int hp, int wp) {
    if (hp <= 0 || hp >= HP - 1 || wp <= 0 || wp >= HP - 1) return 0.0f;
    return src[(hp - 1) * Ww + (wp - 1)];
}

__global__ __launch_bounds__(128) void sampler(const float* __restrict__ inp,
                                               const float* __restrict__ hcur,
                                               const float* __restrict__ moff,
                                               const float* __restrict__ pw,
                                               const float* __restrict__ pb,
                                               unsigned short* __restrict__ Bbf) {
    int pix = blockIdx.x;
    int b = pix >> 12, hw = pix & (HW - 1);
    int h = hw >> 6, w = hw & 63;

    __shared__ float offs[2 * NN];
    __shared__ float gw[4][NN];
    __shared__ int   qx[2][NN], qy[2][NN];

    int tid = threadIdx.x;
    if (tid < 2 * NN) {                                 // 3x3 offset conv, pad 1
        float a = pb[tid];
        const float* mo = moff + (size_t)b * HW;
#pragma unroll
        for (int dh = 0; dh < 3; ++dh)
#pragma unroll
            for (int dw = 0; dw < 3; ++dw) {
                int hh = h + dh - 1, ww2 = w + dw - 1;
                float v = (hh >= 0 && hh < Hh && ww2 >= 0 && ww2 < Ww)
                              ? mo[hh * Ww + ww2] : 0.0f;
                a += v * pw[tid * 9 + dh * 3 + dw];
            }
        offs[tid] = a;
    }
    __syncthreads();
    if (tid < NN) {                                     // bilinear corners/weights
        int n = tid;
        float px = (float)(h + 1) + (float)(n / 3 - 1) + offs[n];
        float py = (float)(w + 1) + (float)(n % 3 - 1) + offs[NN + n];
        float fx = floorf(px), fy = floorf(py);
        int x0 = min(max((int)fx, 0), HP - 1), x1 = min(max((int)fx + 1, 0), HP - 1);
        int y0 = min(max((int)fy, 0), HP - 1), y1 = min(max((int)fy + 1, 0), HP - 1);
        px = fminf(fmaxf(px, 0.0f), (float)(HP - 1));
        py = fminf(fmaxf(py, 0.0f), (float)(HP - 1));
        gw[0][n] = (1.0f + ((float)x0 - px)) * (1.0f + ((float)y0 - py)); // lt
        gw[1][n] = (1.0f - ((float)x1 - px)) * (1.0f - ((float)y1 - py)); // rb
        gw[2][n] = (1.0f + ((float)x0 - px)) * (1.0f - ((float)y1 - py)); // lb
        gw[3][n] = (1.0f - ((float)x1 - px)) * (1.0f + ((float)y0 - py)); // rt
        qx[0][n] = x0; qx[1][n] = x1; qy[0][n] = y0; qy[1][n] = y1;
    }
    __syncthreads();

    int c = tid;                                        // channel of concat(x, h)
    const float* src = (c < CIN) ? (inp  + ((size_t)b * CIN + c)       * HW)
                                 : (hcur + ((size_t)b * HID + (c - CIN)) * HW);
    unsigned short* dst = Bbf + ((size_t)b * KK + (size_t)c * NN) * HW + hw;
#pragma unroll
    for (int n = 0; n < NN; ++n) {
        float v = gw[0][n] * fetch_pad(src, qx[0][n], qy[0][n])
                + gw[1][n] * fetch_pad(src, qx[1][n], qy[1][n])
                + gw[2][n] * fetch_pad(src, qx[0][n], qy[1][n])
                + gw[3][n] * fetch_pad(src, qx[1][n], qy[0][n]);
        dst[(size_t)n * HW] = f2bf(v);
    }
}

// ---------------------------------------------------------------------------
// K5: main WMMA GEMM + fused LSTM epilogue (double-buffered LDS staging).
//   Per block: D(256 x 64) = A(256 x 1152) * B(1152 x 64).
//   Wave w owns M-tiles {g*64 + (w>>1)*16 : g=0..3} and cols (w&1)*32..+32,
//   so the i/f/o/g gate values for each (hid,hw) sit in one wave's registers.
// ---------------------------------------------------------------------------
#define LDA 40   // halves per LDS row (32 data + stagger), 80B => 16B aligned
#define LDB 40

__global__ __launch_bounds__(256) void gemm_lstm(const unsigned short* __restrict__ Abf,
                                                 const unsigned short* __restrict__ Bbf,
                                                 const float* __restrict__ bmv,
                                                 const float* __restrict__ ccur,
                                                 float* __restrict__ out) {
    __shared__ unsigned short Asm[2][OUTC * LDA];       // 2 x 20480 B
    __shared__ unsigned short Bsm[2][64 * LDB];         // 2 x  5120 B

    const int b      = blockIdx.x >> 6;
    const int hwbase = (blockIdx.x & 63) * 64;
    const int tid    = threadIdx.x;
    const int lane   = tid & 31;
    const int wv     = tid >> 5;
    const int wrow   = (wv >> 1) * 16;                  // row base inside gate block
    const int wcol   = (wv & 1) * 32;                   // col base inside strip

    const unsigned short* Ag = Abf + (size_t)b * OUTC * KK;
    const unsigned short* Bg = Bbf + (size_t)b * KK * HW + hwbase;

    v8f acc[4][2];
#pragma unroll
    for (int g = 0; g < 4; ++g)
#pragma unroll
        for (int nt = 0; nt < 2; ++nt)
#pragma unroll
            for (int j = 0; j < 8; ++j) acc[g][nt][j] = 0.0f;

    const int mrow  = lane & 15;
    const int hb    = (lane >> 4) * 8;                  // A lane-half K offset
    const int khalf = (lane >> 4) * 16;                 // unused; B reads also use hb
    const int bkk   = tid & 31;                         // B-staging k row
    const int bng   = tid >> 5;                         // B-staging col group

    // ---- prologue: stage chunk 0 into buffer 0 (synchronous) ----
    {
        const uint4* srcv = (const uint4*)(Ag + (size_t)tid * KK);
        uint4* dstv = (uint4*)(&Asm[0][tid * LDA]);
        dstv[0] = srcv[0]; dstv[1] = srcv[1]; dstv[2] = srcv[2]; dstv[3] = srcv[3];
        union { uint4 v; unsigned short hsz[8]; } t;
        t.v = *(const uint4*)(Bg + (size_t)bkk * HW + bng * 8);
#pragma unroll
        for (int i = 0; i < 8; ++i) Bsm[0][(bng * 8 + i) * LDB + bkk] = t.hsz[i];
    }
    __syncthreads();

    for (int kc = 0; kc < NKC; ++kc) {
        const int cur = kc & 1, nxt = cur ^ 1;
        const int k1  = (kc + 1) * 32;
        const bool pf = (kc + 1 < NKC);

        // ---- prefetch chunk kc+1 while current chunk computes ----
        union { uint4 v; unsigned short hsz[8]; } bpre;
#if !defined(USE_ASYNC_LDS)
        uint4 apre[4];
#endif
        if (pf) {
#if defined(USE_ASYNC_LDS)
            // A slab: async global -> LDS B128, tracked by ASYNCcnt
            const unsigned short* srcA = Ag + (size_t)tid * KK + k1;
#pragma unroll
            for (int i = 0; i < 4; ++i) {
                __builtin_amdgcn_global_load_async_to_lds_b128(
                    (gptr_v4i)(srcA + i * 8),
                    (lptr_v4i)(&Asm[nxt][tid * LDA + i * 8]),
                    0, 0);
            }
#else
            const uint4* srcA = (const uint4*)(Ag + (size_t)tid * KK + k1);
#pragma unroll
            for (int i = 0; i < 4; ++i) apre[i] = srcA[i];
#endif
            bpre.v = *(const uint4*)(Bg + (size_t)(k1 + bkk) * HW + bng * 8);
        }

        // ---- pack fragments (ISA 7.12.2 16-bit layouts) and issue WMMA ----
        v16bf afr[4], bfr[2];
#pragma unroll
        for (int g = 0; g < 4; ++g) {
            int m = g * 64 + wrow + mrow;
            union { v16bf v; unsigned u[8]; } t;
#pragma unroll
            for (int j = 0; j < 8; ++j) {
                int k = ((j >> 2) * 16) + hb + (j & 3) * 2;   // K pair base
                t.u[j] = *(const unsigned*)&Asm[cur][m * LDA + k];
            }
            afr[g] = t.v;
        }
#pragma unroll
        for (int nt = 0; nt < 2; ++nt) {
            int n = wcol + nt * 16 + (lane & 15);
            union { v16bf v; unsigned u[8]; } t;
#pragma unroll
            for (int j = 0; j < 8; ++j)
                t.u[j] = *(const unsigned*)&Bsm[cur][n * LDB + ((j >> 2) * 16) + hb + (j & 3) * 2];
            bfr[nt] = t.v;
        }
#pragma unroll
        for (int g = 0; g < 4; ++g)
#pragma unroll
            for (int nt = 0; nt < 2; ++nt)
                acc[g][nt] = __builtin_amdgcn_wmma_f32_16x16x32_bf16(
                    false, afr[g], false, bfr[nt], (short)0, acc[g][nt], false, false);

        // ---- commit prefetched chunk into the other buffer ----
        if (pf) {
#if !defined(USE_ASYNC_LDS)
            uint4* dstv = (uint4*)(&Asm[nxt][tid * LDA]);
#pragma unroll
            for (int i = 0; i < 4; ++i) dstv[i] = apre[i];
#endif
#pragma unroll
            for (int i = 0; i < 8; ++i) Bsm[nxt][(bng * 8 + i) * LDB + bkk] = bpre.hsz[i];
        }
#if defined(USE_ASYNC_LDS)
        if (pf) __builtin_amdgcn_s_wait_asynccnt(0);
#endif
        __syncthreads();
    }

    // ---- fused LSTM epilogue: gates for each (hid,hw) are in-register ----
    const float* bmb = bmv + b * OUTC;
#pragma unroll
    for (int j = 0; j < 8; ++j) {
        int lr  = j + ((lane >> 4) << 3);               // local C/D row
        int hid = wrow + lr;                            // 0..63 within gate block
        float bi = bmb[hid], bf = bmb[64 + hid], bo = bmb[128 + hid], bg = bmb[192 + hid];
#pragma unroll
        for (int nt = 0; nt < 2; ++nt) {
            int hw = hwbase + wcol + nt * 16 + (lane & 15);
            float iv = sigmoidf(acc[0][nt][j] + bi);
            float fv = sigmoidf(acc[1][nt][j] + bf);
            float ov = sigmoidf(acc[2][nt][j] + bo);
            float gv = tanhf(acc[3][nt][j] + bg);
            size_t base = ((size_t)b * HID + hid) * HW + hw;
            float cp = ccur[base];
            float cn = fv * cp + iv * gv;
            float hn = ov * tanhf(cn);
            out[base] = hn;                             // h_next
            out[(size_t)Bm * HID * HW + base] = cn;     // c_next
        }
    }
}

// ---------------------------------------------------------------------------
// host launcher
// ---------------------------------------------------------------------------
extern "C" void kernel_launch(void* const* d_in, const int* in_sizes, int n_in,
                              void* d_out, int out_size, void* d_ws, size_t ws_size,
                              hipStream_t stream) {
    const float* input = (const float*)d_in[0];
    const float* hcur  = (const float*)d_in[1];
    const float* ccur  = (const float*)d_in[2];
    const float* meta  = (const float*)d_in[3];
    const float* moff  = (const float*)d_in[4];
    const float* w1w   = (const float*)d_in[5];
    const float* w1b   = (const float*)d_in[6];
    const float* w2w   = (const float*)d_in[7];
    const float* w2b   = (const float*)d_in[8];
    const float* blw   = (const float*)d_in[9];
    const float* blb   = (const float*)d_in[10];
    const float* pw    = (const float*)d_in[11];
    const float* pb    = (const float*)d_in[12];
    float* out = (float*)d_out;

    char* ws = (char*)d_ws;                             // ~40.6 MB used
    float*          wm1 = (float*)(ws + 0);             //  524288 B
    unsigned short* Abf = (unsigned short*)(ws + 524288);   // 2359296 B
    float*          bmv = (float*)(ws + 2883584);       //    4096 B
    unsigned short* Bbf = (unsigned short*)(ws + 2887680);  // 37748736 B

    hyper_w1  <<<  512, 256, 0, stream>>>(meta, w1w, w1b, wm1);
    hyper_w2  <<< 4608, 256, 0, stream>>>(wm1, w2w, w2b, Abf);
    hyper_bias<<<    4, 256, 0, stream>>>(meta, blw, blb, bmv);
    sampler   <<<16384, 128, 0, stream>>>(input, hcur, moff, pw, pb, Bbf);
    gemm_lstm <<<  256, 256, 0, stream>>>(Abf, Bbf, bmv, ccur, out);
}
